// MinkUNet_63496796504737
// MI455X (gfx1250) — matrix operand order repaired and checked by reading, not compiled, not run on hardware
//
#include <hip/hip_runtime.h>
#include <vector>
#include <string>
#include <algorithm>
#include <map>

// ---------------------------------------------------------------------------
// Types for CDNA5 WMMA (wave32): v_wmma_f32_16x16x32_bf16
// ---------------------------------------------------------------------------
typedef __attribute__((ext_vector_type(16))) __bf16          v16bf;
typedef __attribute__((ext_vector_type(8)))  float           v8f;
typedef __attribute__((ext_vector_type(16))) unsigned short  u16x16;
typedef __attribute__((ext_vector_type(8)))  unsigned short  u16x8;

__device__ __forceinline__ unsigned short f2bf(float f) {
    unsigned u = __builtin_bit_cast(unsigned, f);
    return (unsigned short)((u + 0x7FFFu + ((u >> 16) & 1u)) >> 16);  // RNE
}

__device__ __forceinline__ v8f zero8() {
    v8f z;
#pragma unroll
    for (int i = 0; i < 8; ++i) z[i] = 0.f;
    return z;
}

// Async copy of 32 bytes global -> LDS per lane (2 x b128, shared ASYNCcnt).
// offset:16 applies to both the LDS and global side per the TDM/async spec.
__device__ __forceinline__ void stage_async32(const unsigned short* g, unsigned short* l) {
    unsigned lds = (unsigned)(size_t)l;      // low 32 bits of generic ptr = LDS offset
    asm volatile("global_load_async_to_lds_b128 %0, %1, off\n\t"
                 "global_load_async_to_lds_b128 %0, %1, off offset:16"
                 :: "v"(lds), "v"(g) : "memory");
}
__device__ __forceinline__ void wait_async0() {
    asm volatile("s_wait_asynccnt 0" ::: "memory");
}

// ---------------------------------------------------------------------------
// f32 -> bf16 mirror (packed pair per thread). Done once per conv source so
// the conv inner loop does zero conversion work and reads half the bytes.
// ---------------------------------------------------------------------------
__global__ void k_tobf16(const float* __restrict__ src, unsigned int* __restrict__ dst,
                         long long npairs)
{
    const long long id = (long long)blockIdx.x * 256 + threadIdx.x;
    if (id >= npairs) return;
    const float a = src[2 * id], b = src[2 * id + 1];
    dst[id] = (unsigned)f2bf(a) | ((unsigned)f2bf(b) << 16);
}

// ---------------------------------------------------------------------------
// Implicit-GEMM conv (3x3x3 or 1x1x1), dense masked channels-last grid.
// Block = 4 waves; wave = 16 voxels (M) x 64 out-channels (4 WMMA N-tiles).
// B tiles (4KB each) double-buffered in LDS via async-to-LDS loads; one
// workgroup barrier per (tap,kchunk) iteration.
// A: two 16B bf16 vector loads per lane per iteration.
// Epilogue: +bias, *mask, strided store (concat offset / NKDHW out).
// ---------------------------------------------------------------------------
__global__ __launch_bounds__(128) void k_conv_wmma(
    const unsigned short* __restrict__ src, int srcC,      // bf16 mirror
    const unsigned short* __restrict__ wpack,
    const float* __restrict__ bias,                        // padded to COp or null
    const float* __restrict__ mask,
    float* __restrict__ dst, long long voxStride, long long chanStride, long long dstOff,
    int D, int Db, int CIp, int COp, int COreal, int taps)
{
    __shared__ unsigned short Bs[4096];                    // 2 x (4 n-tiles x 512)

    const int lane = threadIdx.x & 31;
    const int wave = threadIdx.x >> 5;
    const int half = lane >> 4;       // K-half for A, row-group for D
    const int row  = lane & 15;       // M index for A, N index for B/D
    const int vtile = (blockIdx.x * 4 + wave) * 16;
    const int v = vtile + row;
    const int Dm = D - 1;
    const int x = v & Dm, y = (v >> Db) & Dm, z = v >> (2 * Db);
    const int n0 = blockIdx.y * 64;
    const int KC    = CIp >> 5;
    const int NTtot = COp >> 4;
    const int nt0   = n0 >> 4;
    const int NIT   = taps * KC;
    const int tid16 = threadIdx.x * 16;

    v8f acc0 = zero8(), acc1 = zero8(), acc2 = zero8(), acc3 = zero8();

    // prologue: stage iteration 0 (weight layout is contiguous in (t,kc))
    stage_async32(wpack + (size_t)nt0 * 512 + tid16, Bs + tid16);

    int t = 0, kc = 0;
    bool valid = false;
    const unsigned short* sp0 = src;

    for (int it = 0; it < NIT; ++it) {
        if (kc == 0) {                       // new tap: recompute neighbor base
            int dz = 0, dy = 0, dx = 0;
            if (taps == 27) { dz = t / 9 - 1; dy = (t / 3) % 3 - 1; dx = t % 3 - 1; }
            const int nx = x + dx, ny = y + dy, nz = z + dz;
            valid = (nx >= 0) & (nx < D) & (ny >= 0) & (ny < D) & (nz >= 0) & (nz < D);
            const long long nb = (long long)(nz * D + ny) * D + nx;
            sp0 = src + nb * srcC + half * 8;
        }

        wait_async0();                       // my wave's staged tile arrived
        __syncthreads();                     // all waves' tiles arrived; prev reads done
        const int cur = it & 1;
        if (it + 1 < NIT)
            stage_async32(wpack + ((size_t)(it + 1) * NTtot + nt0) * 512 + tid16,
                          Bs + ((it + 1) & 1) * 2048 + tid16);

        // ---- per-lane A fragment (bf16, vector loads) ----
        u16x16 a;
        if (valid) {
            const unsigned short* p = sp0 + kc * 32;
            const u16x8 q0 = *(const u16x8*)(p);
            const u16x8 q1 = *(const u16x8*)(p + 16);
#pragma unroll
            for (int e = 0; e < 8; ++e) { a[e] = q0[e]; a[e + 8] = q1[e]; }
        } else {
#pragma unroll
            for (int e = 0; e < 16; ++e) a[e] = 0;
        }
        const v16bf av = __builtin_bit_cast(v16bf, a);

        const u16x16* bsp = (const u16x16*)(Bs + cur * 2048);
        acc0 = __builtin_amdgcn_wmma_f32_16x16x32_bf16(false, av, false,
                  __builtin_bit_cast(v16bf, bsp[lane]),      (short)0, acc0, false, false);
        acc1 = __builtin_amdgcn_wmma_f32_16x16x32_bf16(false, av, false,
                  __builtin_bit_cast(v16bf, bsp[32 + lane]), (short)0, acc1, false, false);
        acc2 = __builtin_amdgcn_wmma_f32_16x16x32_bf16(false, av, false,
                  __builtin_bit_cast(v16bf, bsp[64 + lane]), (short)0, acc2, false, false);
        acc3 = __builtin_amdgcn_wmma_f32_16x16x32_bf16(false, av, false,
                  __builtin_bit_cast(v16bf, bsp[96 + lane]), (short)0, acc3, false, false);

        if (++kc == KC) { kc = 0; ++t; }
    }

    float b0 = 0.f, b1 = 0.f, b2 = 0.f, b3 = 0.f;
    if (bias) {
        b0 = bias[n0 + row];      b1 = bias[n0 + 16 + row];
        b2 = bias[n0 + 32 + row]; b3 = bias[n0 + 48 + row];
    }
#pragma unroll
    for (int j = 0; j < 8; ++j) {
        const int vv = vtile + half * 8 + j;                 // D row M = j + 8*half
        const float mm = mask[vv];
        const long long bb = (long long)vv * voxStride + dstOff;
        int n;
        n = n0 + row;      if (n < COreal) dst[bb + (long long)n * chanStride] = (acc0[j] + b0) * mm;
        n = n0 + 16 + row; if (n < COreal) dst[bb + (long long)n * chanStride] = (acc1[j] + b1) * mm;
        n = n0 + 32 + row; if (n < COreal) dst[bb + (long long)n * chanStride] = (acc2[j] + b2) * mm;
        n = n0 + 48 + row; if (n < COreal) dst[bb + (long long)n * chanStride] = (acc3[j] + b3) * mm;
    }
}

// ---------------------------------------------------------------------------
// Repack (CO,CI,k,k,k) f32 weights -> per-lane bf16 WMMA-B layout
// [tap][kchunk][ntile][lane][16]; zero-pad CI->CIp, CO->COp; pad bias.
// ---------------------------------------------------------------------------
__global__ void k_repack(const float* __restrict__ w, const float* __restrict__ b,
                         unsigned short* __restrict__ wpack, float* __restrict__ bpad,
                         int taps, int CI, int CO, int CIp, int COp)
{
    const int NTt = COp >> 4, KC = CIp >> 5;
    const size_t total = (size_t)taps * KC * NTt * 512;
    for (size_t idx = (size_t)blockIdx.x * 256 + threadIdx.x; idx < total;
         idx += (size_t)gridDim.x * 256) {
        const int e    = (int)(idx & 15);
        const int lane = (int)((idx >> 4) & 31);
        size_t rest = idx >> 9;
        const int ntA = (int)(rest % NTt); rest /= NTt;
        const int kc  = (int)(rest % KC);
        const int t   = (int)(rest / KC);
        const int n = ntA * 16 + (lane & 15);
        const int k = kc * 32 + (lane >> 4) * 8 + (e & 7) + (e >> 3) * 16;
        float val = 0.f;
        if (k < CI && n < CO) val = w[((size_t)n * CI + k) * taps + t];
        wpack[idx] = f2bf(val);
    }
    if (b && blockIdx.x == 0 && (int)threadIdx.x < COp)
        bpad[threadIdx.x] = ((int)threadIdx.x < CO) ? b[threadIdx.x] : 0.f;
}

// ---------------------------------------------------------------------------
// Input prep: mask m0 from occ, pad feats (NCDHW, 4ch) -> channels-last 32ch.
// ---------------------------------------------------------------------------
__global__ void k_prep0(const float* __restrict__ feats, const unsigned char* __restrict__ occ,
                        float* __restrict__ xin32, float* __restrict__ m0,
                        float* __restrict__ cnt, int nvox)
{
    __shared__ float sc;
    if (threadIdx.x == 0) sc = 0.f;
    __syncthreads();
    const int v = blockIdx.x * 256 + threadIdx.x;
    float m = 0.f;
    if (v < nvox) {
        m = occ[v] ? 1.f : 0.f;
        m0[v] = m;
#pragma unroll
        for (int c = 0; c < 32; ++c)
            xin32[(size_t)v * 32 + c] = (c < 4 && m > 0.f) ? feats[(size_t)c * nvox + v] : 0.f;
    }
    atomicAdd(&sc, m);
    __syncthreads();
    if (threadIdx.x == 0) atomicAdd(cnt, sc);
}

__global__ void k_poolmask(const float* __restrict__ mf, float* __restrict__ mc,
                           float* __restrict__ cnt, int Dc, int Dcb)
{
    __shared__ float sc;
    if (threadIdx.x == 0) sc = 0.f;
    __syncthreads();
    const int nvc = Dc * Dc * Dc;
    const int v = blockIdx.x * 256 + threadIdx.x;
    float m = 0.f;
    if (v < nvc) {
        const int Dm = Dc - 1;
        const int x = v & Dm, y = (v >> Dcb) & Dm, z = v >> (2 * Dcb);
        const int Df = 2 * Dc;
#pragma unroll
        for (int i = 0; i < 8; ++i) {
            const int fx = 2 * x + (i & 1), fy = 2 * y + ((i >> 1) & 1), fz = 2 * z + (i >> 2);
            m = fmaxf(m, mf[((size_t)fz * Df + fy) * Df + fx]);
        }
        mc[v] = m;
    }
    atomicAdd(&sc, m);
    __syncthreads();
    if (threadIdx.x == 0) atomicAdd(cnt, sc);
}

__global__ void k_maxpool(const float* __restrict__ xf, const float* __restrict__ mf,
                          const float* __restrict__ mc, float* __restrict__ xc,
                          int Dc, int Dcb, int C, int Cb)
{
    const long long id = (long long)blockIdx.x * 256 + threadIdx.x;
    const long long nvc = (long long)Dc * Dc * Dc;
    if (id >= (nvc << Cb)) return;
    const int c = (int)(id & (C - 1));
    const int v = (int)(id >> Cb);
    const int Dm = Dc - 1;
    const int x = v & Dm, y = (v >> Dcb) & Dm, z = v >> (2 * Dcb);
    const int Df = 2 * Dc;
    float best = -3.0e38f;
#pragma unroll
    for (int i = 0; i < 8; ++i) {
        const int fx = 2 * x + (i & 1), fy = 2 * y + ((i >> 1) & 1), fz = 2 * z + (i >> 2);
        const long long fv = ((long long)fz * Df + fy) * Df + fx;
        if (mf[fv] > 0.f) best = fmaxf(best, xf[(fv << Cb) + c]);
    }
    xc[((long long)v << Cb) + c] = (mc[v] > 0.f) ? best : 0.f;
}

__global__ void k_upsample(const float* __restrict__ xc, const float* __restrict__ mf,
                           float* __restrict__ dst, long long dstStride, long long dstOff,
                           int Df, int Dfb, int C, int Cb)
{
    const long long id = (long long)blockIdx.x * 256 + threadIdx.x;
    const long long nv = (long long)Df * Df * Df;
    if (id >= (nv << Cb)) return;
    const int c = (int)(id & (C - 1));
    const long long v = id >> Cb;
    const int Dm = Df - 1;
    const int x = (int)v & Dm, y = ((int)(v >> Dfb)) & Dm, z = (int)(v >> (2 * Dfb));
    const int Dc = Df >> 1;
    const long long pv = ((long long)(z >> 1) * Dc + (y >> 1)) * Dc + (x >> 1);
    dst[v * dstStride + dstOff + c] = xc[(pv << Cb) + c] * mf[v];
}

__global__ void k_copycat(const float* __restrict__ src, int srcC, float* __restrict__ dst,
                          long long dstStride, long long dstOff, int nvox, int C, int Cb)
{
    const long long id = (long long)blockIdx.x * 256 + threadIdx.x;
    if (id >= ((long long)nvox << Cb)) return;
    const int c = (int)(id & (C - 1));
    const long long v = id >> Cb;
    dst[v * dstStride + dstOff + c] = src[v * srcC + c];
}

// ---------------------------------------------------------------------------
// Sparse-BN: per-channel sum / sumsq (input already zero at inactive voxels),
// finalize to scale/shift, fused apply (+residual, ReLU, mask).
// ---------------------------------------------------------------------------
__global__ void k_bn_stats(const float* __restrict__ t, float* __restrict__ sums,
                           int C, int Cb, int nvox)
{
    const int c   = threadIdx.x & (C - 1);
    const int sub = threadIdx.x >> Cb;
    const int nw  = 256 >> Cb;
    float s = 0.f, q = 0.f;
    for (int v = blockIdx.x * nw + sub; v < nvox; v += gridDim.x * nw) {
        const float x = t[((size_t)v << Cb) + c];
        s += x; q += x * x;
    }
    atomicAdd(&sums[c], s);
    atomicAdd(&sums[256 + c], q);
}

__global__ void k_bn_final(const float* __restrict__ sums, const float* __restrict__ cnt,
                           const float* __restrict__ g, const float* __restrict__ be,
                           float* __restrict__ scale, float* __restrict__ shift, int C)
{
    const int c = threadIdx.x;
    if (c >= C) return;
    const float n   = fmaxf(cnt[0], 1.f);
    const float mu  = sums[c] / n;
    const float var = sums[256 + c] / n - mu * mu;
    const float rs  = rsqrtf(var + 1e-5f);
    scale[c] = g[c] * rs;
    shift[c] = be[c] - mu * g[c] * rs;
}

__global__ void k_bn_apply(const float* __restrict__ t, const float* __restrict__ res,
                           const float* __restrict__ mask, const float* __restrict__ scale,
                           const float* __restrict__ shift, float* __restrict__ dst,
                           int C, int Cb, int nvox, int relu)
{
    const long long id = (long long)blockIdx.x * 256 + threadIdx.x;
    if (id >= ((long long)nvox << Cb)) return;
    const int c = (int)(id & (C - 1));
    const long long v = id >> Cb;
    float val = scale[c] * t[id] + shift[c];
    if (res)  val += res[id];
    if (relu) val = fmaxf(val, 0.f);
    dst[id] = val * mask[v];
}

// ---------------------------------------------------------------------------
// Host orchestration
// ---------------------------------------------------------------------------
struct BlkDef { const char* n; int ci, co; };
static const BlkDef g_blocks[16] = {
    {"b2a",64,64},{"b2b",64,64},{"b4a",64,64},{"b4b",64,64},
    {"b8a",64,128},{"b8b",128,128},{"b16a",128,256},{"b16b",256,256},
    {"b8c",384,256},{"b8d",256,256},{"b4d",192,128},{"b4e",128,128},
    {"b2d",128,64},{"b2e",64,64},{"b1d",128,64},{"b1e",64,64}};

static inline int ilog2i(int v) { int b = 0; while ((1 << b) < v) ++b; return b; }

extern "C" void kernel_launch(void* const* d_in, const int* in_sizes, int n_in,
                              void* d_out, int out_size, void* d_ws, size_t ws_size,
                              hipStream_t stream)
{
    (void)in_sizes; (void)n_in; (void)out_size; (void)ws_size;

    // --- param-name table (jax pytree flattens dict leaves sorted by key) ---
    std::vector<std::string> names;
    auto cwn = [&](const std::string& n, bool bias) { names.push_back(n + "_w"); if (bias) names.push_back(n + "_b"); };
    auto bnn = [&](const std::string& n) { names.push_back(n + "_g"); names.push_back(n + "_be"); };
    cwn("stem1", true); bnn("stembn1"); cwn("stem2", true); bnn("stembn2");
    for (int i = 0; i < 16; ++i) {
        const std::string n = g_blocks[i].n;
        cwn(n + "_c1", true); bnn(n + "_bn1");
        cwn(n + "_c2", true); bnn(n + "_bn2");
        if (g_blocks[i].ci != g_blocks[i].co) { cwn(n + "_d", true); bnn(n + "_bnd"); }
    }
    cwn("up2c", false); cwn("up3c", false); cwn("up4c", false);
    names.push_back("cls_w"); names.push_back("cls_b");
    std::sort(names.begin(), names.end());
    std::map<std::string, const float*> P;
    for (size_t i = 0; i < names.size(); ++i) P[names[i]] = (const float*)d_in[2 + i];

    const float* feats = (const float*)d_in[0];
    const unsigned char* occ = (const unsigned char*)d_in[1];

    // --- arena ---
    char* base = (char*)d_ws;
    size_t off = 0;
    auto A  = [&](size_t bytes) -> void* { void* p = base + off; off = (off + bytes + 255) & ~(size_t)255; return p; };
    auto AF = [&](size_t elems) -> float* { return (float*)A(elems * sizeof(float)); };

    const int Dd[5] = {64, 32, 16, 8, 4};
    const int Db[5] = {6, 5, 4, 3, 2};
    const int NV[5] = {262144, 32768, 4096, 512, 64};

    float* m[5]; for (int l = 0; l < 5; ++l) m[l] = AF(NV[l]);
    float* cnt    = AF(8);
    float* sums   = AF(512);
    float* scaleB = AF(256);
    float* shiftB = AF(256);
    unsigned short* wpackBuf = (unsigned short*)A(6u * 1024 * 1024);
    float* bpadBuf = AF(256);
    unsigned short* srcbf = (unsigned short*)A((size_t)NV[0] * 128 * sizeof(unsigned short));

    // level 0 (aliased lifetimes: xin32<->z0, u0<->tC0)
    float* z0  = AF((size_t)NV[0] * 64);  float* xin32 = z0;
    float* x0  = AF((size_t)NV[0] * 64);
    float* tA0 = AF((size_t)NV[0] * 64);
    float* tB0 = AF((size_t)NV[0] * 64);
    float* tC0 = AF((size_t)NV[0] * 64);  float* u0 = tC0;
    float* c0  = AF((size_t)NV[0] * 128);
    // level 1
    float* x1f = AF((size_t)NV[1] * 64);  float* x1p = AF((size_t)NV[1] * 64);
    float* tA1 = AF((size_t)NV[1] * 64);  float* tB1 = AF((size_t)NV[1] * 64);
    float* tC1 = AF((size_t)NV[1] * 64);
    float* u1  = AF((size_t)NV[1] * 128); float* c1  = AF((size_t)NV[1] * 128);
    float* y1f = AF((size_t)NV[1] * 64);
    // level 2
    float* x2f = AF((size_t)NV[2] * 64);  float* x2p = AF((size_t)NV[2] * 64);
    float* tA2 = AF((size_t)NV[2] * 128); float* tB2 = AF((size_t)NV[2] * 128);
    float* tC2 = AF((size_t)NV[2] * 128);
    float* u2  = AF((size_t)NV[2] * 256); float* c2  = AF((size_t)NV[2] * 192);
    float* y2f = AF((size_t)NV[2] * 128);
    // level 3
    float* x3f = AF((size_t)NV[3] * 128); float* x3p = AF((size_t)NV[3] * 64);
    float* tA3 = AF((size_t)NV[3] * 256); float* tB3 = AF((size_t)NV[3] * 256);
    float* tC3 = AF((size_t)NV[3] * 256);
    float* c3  = AF((size_t)NV[3] * 384); float* y3f = AF((size_t)NV[3] * 256);
    // level 4
    float* x4  = AF((size_t)NV[4] * 256); float* x4p = AF((size_t)NV[4] * 128);
    float* tA4 = AF((size_t)NV[4] * 256); float* tB4 = AF((size_t)NV[4] * 256);
    float* tC4 = AF((size_t)NV[4] * 256);

    auto ew = [&](long long tot) { return dim3((unsigned)((tot + 255) / 256)); };

    // --- helpers ---
    auto conv = [&](const std::string& wname, const char* bname, int taps, int CI, int CO,
                    const float* src, int srcC, const float* mask, int lvl,
                    float* dst, long long voxStride, long long chanStride, long long dstOff,
                    int COreal) {
        const int D = Dd[lvl], nvox = NV[lvl];
        const int CIp = (CI + 31) & ~31;
        const int COp = (CO + 63) & ~63;
        const size_t total = (size_t)taps * (CIp / 32) * (COp / 16) * 512;
        const int rblocks = (int)std::min<size_t>((total + 255) / 256, 4096);
        const float* bb = bname ? P[bname] : nullptr;
        k_repack<<<dim3(rblocks), dim3(256), 0, stream>>>(
            P[wname], bb, wpackBuf, bpadBuf, taps, CI, CO, CIp, COp);
        const long long npairs = (long long)nvox * srcC / 2;
        k_tobf16<<<ew(npairs), dim3(256), 0, stream>>>(src, (unsigned int*)srcbf, npairs);
        dim3 g(nvox / 64, COp / 64);
        k_conv_wmma<<<g, dim3(128), 0, stream>>>(
            srcbf, srcC, wpackBuf, bb ? bpadBuf : nullptr, mask,
            dst, voxStride, chanStride, dstOff, D, Db[lvl], CIp, COp, COreal, taps);
    };

    auto bnrun = [&](const std::string& gn, const std::string& ben, float* t, const float* res,
                     const float* mask, int C, int lvl, int relu, float* dst) {
        const int nvox = NV[lvl], Cb = ilog2i(C);
        hipMemsetAsync(sums, 0, 512 * sizeof(float), stream);
        const int nw = 256 / C;
        const int sblocks = std::max(1, std::min((nvox + nw - 1) / nw, 240));
        k_bn_stats<<<dim3(sblocks), dim3(256), 0, stream>>>(t, sums, C, Cb, nvox);
        k_bn_final<<<dim3(1), dim3(256), 0, stream>>>(sums, cnt + lvl, P[gn], P[ben], scaleB, shiftB, C);
        const long long tot = (long long)nvox * C;
        k_bn_apply<<<ew(tot), dim3(256), 0, stream>>>(
            t, res, mask, scaleB, shiftB, dst, C, Cb, nvox, relu);
    };

    auto resblock = [&](const std::string& n, int ci, int co, float* xin, const float* mask,
                        int lvl, float* tA, float* tB, float* tC, float* out) {
        conv(n + "_c1_w", (n + "_c1_b").c_str(), 27, ci, co, xin, ci, mask, lvl, tA, co, 1, 0, co);
        bnrun(n + "_bn1_g", n + "_bn1_be", tA, nullptr, mask, co, lvl, 1, tA);
        conv(n + "_c2_w", (n + "_c2_b").c_str(), 27, co, co, tA, co, mask, lvl, tB, co, 1, 0, co);
        const float* res;
        if (ci != co) {
            conv(n + "_d_w", (n + "_d_b").c_str(), 1, ci, co, xin, ci, mask, lvl, tC, co, 1, 0, co);
            bnrun(n + "_bnd_g", n + "_bnd_be", tC, nullptr, mask, co, lvl, 0, tC);
            res = tC;
        } else res = xin;
        bnrun(n + "_bn2_g", n + "_bn2_be", tB, res, mask, co, lvl, 1, out);
    };

    // --- masks + input prep ---
    hipMemsetAsync(cnt, 0, 8 * sizeof(float), stream);
    k_prep0<<<ew(NV[0]), dim3(256), 0, stream>>>(feats, occ, xin32, m[0], cnt + 0, NV[0]);
    for (int l = 1; l < 5; ++l)
        k_poolmask<<<ew(NV[l]), dim3(256), 0, stream>>>(m[l - 1], m[l], cnt + l, Dd[l], Db[l]);

    // --- stem (level 0) ---
    conv("stem1_w", "stem1_b", 27, 4, 64, xin32, 32, m[0], 0, tA0, 64, 1, 0, 64);
    bnrun("stembn1_g", "stembn1_be", tA0, nullptr, m[0], 64, 0, 1, tA0);
    conv("stem2_w", "stem2_b", 27, 64, 64, tA0, 64, m[0], 0, tB0, 64, 1, 0, 64);
    bnrun("stembn2_g", "stembn2_be", tB0, nullptr, m[0], 64, 0, 1, x0);

    // --- encoder ---
    k_maxpool<<<ew((long long)NV[1] * 64), dim3(256), 0, stream>>>(x0, m[0], m[1], x1p, Dd[1], Db[1], 64, 6);
    resblock("b2a", 64, 64, x1p, m[1], 1, tA1, tB1, tC1, x1f);
    resblock("b2b", 64, 64, x1f, m[1], 1, tA1, tB1, tC1, x1f);

    k_maxpool<<<ew((long long)NV[2] * 64), dim3(256), 0, stream>>>(x1f, m[1], m[2], x2p, Dd[2], Db[2], 64, 6);
    resblock("b4a", 64, 64, x2p, m[2], 2, tA2, tB2, tC2, x2f);
    resblock("b4b", 64, 64, x2f, m[2], 2, tA2, tB2, tC2, x2f);

    k_maxpool<<<ew((long long)NV[3] * 64), dim3(256), 0, stream>>>(x2f, m[2], m[3], x3p, Dd[3], Db[3], 64, 6);
    resblock("b8a", 64, 128, x3p, m[3], 3, tA3, tB3, tC3, x3f);
    resblock("b8b", 128, 128, x3f, m[3], 3, tA3, tB3, tC3, x3f);

    k_maxpool<<<ew((long long)NV[4] * 128), dim3(256), 0, stream>>>(x3f, m[3], m[4], x4p, Dd[4], Db[4], 128, 7);
    resblock("b16a", 128, 256, x4p, m[4], 4, tA4, tB4, tC4, x4);
    resblock("b16b", 256, 256, x4, m[4], 4, tA4, tB4, tC4, x4);

    // --- decoder ---
    k_upsample<<<ew((long long)NV[3] * 256), dim3(256), 0, stream>>>(x4, m[3], c3, 384, 0, Dd[3], Db[3], 256, 8);
    k_copycat<<<ew((long long)NV[3] * 128), dim3(256), 0, stream>>>(x3f, 128, c3, 384, 256, NV[3], 128, 7);
    resblock("b8c", 384, 256, c3, m[3], 3, tA3, tB3, tC3, y3f);
    resblock("b8d", 256, 256, y3f, m[3], 3, tA3, tB3, tC3, y3f);

    k_upsample<<<ew((long long)NV[2] * 256), dim3(256), 0, stream>>>(y3f, m[2], u2, 256, 0, Dd[2], Db[2], 256, 8);
    conv("up2c_w", nullptr, 1, 256, 128, u2, 256, m[2], 2, c2, 192, 1, 0, 128);
    k_copycat<<<ew((long long)NV[2] * 64), dim3(256), 0, stream>>>(x2f, 64, c2, 192, 128, NV[2], 64, 6);
    resblock("b4d", 192, 128, c2, m[2], 2, tA2, tB2, tC2, y2f);
    resblock("b4e", 128, 128, y2f, m[2], 2, tA2, tB2, tC2, y2f);

    k_upsample<<<ew((long long)NV[1] * 128), dim3(256), 0, stream>>>(y2f, m[1], u1, 128, 0, Dd[1], Db[1], 128, 7);
    conv("up3c_w", nullptr, 1, 128, 64, u1, 128, m[1], 1, c1, 128, 1, 0, 64);
    k_copycat<<<ew((long long)NV[1] * 64), dim3(256), 0, stream>>>(x1f, 64, c1, 128, 64, NV[1], 64, 6);
    resblock("b2d", 128, 64, c1, m[1], 1, tA1, tB1, tC1, y1f);
    resblock("b2e", 64, 64, y1f, m[1], 1, tA1, tB1, tC1, y1f);

    k_upsample<<<ew((long long)NV[0] * 64), dim3(256), 0, stream>>>(y1f, m[0], u0, 64, 0, Dd[0], Db[0], 64, 6);
    conv("up4c_w", nullptr, 1, 64, 64, u0, 64, m[0], 0, c0, 128, 1, 0, 64);
    k_copycat<<<ew((long long)NV[0] * 64), dim3(256), 0, stream>>>(x0, 64, c0, 128, 64, NV[0], 64, 6);
    resblock("b1d", 128, 64, c0, m[0], 0, tA0, tB0, tC0, z0);
    resblock("b1e", 64, 64, z0, m[0], 0, tA0, tB0, tC0, z0);

    // --- classifier head: 1x1 GEMM, direct NKDHW store to d_out (masked) ---
    conv("cls_w", "cls_b", 1, 64, 50, z0, 64, m[0], 0, (float*)d_out, 1, NV[0], 0, 50);
}
